// Attention_61821759259143
// MI455X (gfx1250) — matrix operand order repaired and checked by reading, not compiled, not run on hardware
//
#include <hip/hip_runtime.h>
#include <hip/hip_bf16.h>

typedef _Float16 half_t;
typedef __attribute__((ext_vector_type(16))) _Float16 v16h;
typedef __attribute__((ext_vector_type(8)))  _Float16 v8h;
typedef __attribute__((ext_vector_type(8)))  float    v8f;
typedef __attribute__((ext_vector_type(4)))  float    v4f;
typedef __attribute__((ext_vector_type(4)))  int      v4i;

union Frag {
    v16h h;
    _Float16 e[16];
};

__device__ __forceinline__ v8f wmma_f16(v16h a, v16h b, v8f c) {
    return __builtin_amdgcn_wmma_f32_16x16x32_f16(false, a, false, b, (short)0, c, false, false);
}

// ---------------- CDNA5 async global->LDS copy (ASYNCcnt path) ----------
#ifndef __has_builtin
#define __has_builtin(x) 0
#endif
#if __has_builtin(__builtin_amdgcn_global_load_async_to_lds_b128)
#define HAVE_ASYNC 1
#else
#define HAVE_ASYNC 0
#endif

#if HAVE_ASYNC
typedef __attribute__((address_space(1))) _Float16 glob_half;
typedef __attribute__((address_space(3))) _Float16 lds_half;
typedef __attribute__((address_space(1))) v4i      glob_v4i;
typedef __attribute__((address_space(3))) v4i      lds_v4i;
#endif

// 16-byte (8-half) async copy, global -> LDS
__device__ __forceinline__ void async_cp16(_Float16* lds, const half_t* g) {
#if HAVE_ASYNC
    __builtin_amdgcn_global_load_async_to_lds_b128(
        (glob_v4i*)(glob_half*)g, (lds_v4i*)(lds_half*)lds, 0, 0);
#else
    *(v8h*)lds = *(const v8h*)g;   // fallback: through VGPRs
#endif
}

template <int N>
__device__ __forceinline__ void wait_async() {
#if HAVE_ASYNC
# if __has_builtin(__builtin_amdgcn_s_wait_asynccnt)
    __builtin_amdgcn_s_wait_asynccnt(N);
# else
    asm volatile("s_wait_asynccnt %0" :: "i"(N) : "memory");
# endif
#endif
}

// ---------------- problem constants ----------------
#define Bb   2
#define Nn   2048
#define Cc   768
#define Hh   12
#define HD   64
#define Mm   (Bb * Nn)            // 4096
#define TOT  (Bb * Hh * Nn * HD)  // 3145728 halves per Q/K/V/O matrix
#define WSZ  (Cc * Cc)            // 589824 halves per weight matrix

// ---------------- GEMM tiling ----------------
#define BM 128
#define BN 64
#define BK 32
#define LDA_STR 40   // halves; 80 B = 5*16 -> b128-aligned rows
#define LDB_STR 40
#define NKT (Cc / BK) // 24

// ======================================================================
// Kernel 0: f32 -> f16 convert (x and the four weight matrices).
// ======================================================================
__global__ __launch_bounds__(256) void cvt_kernel(
    const float* __restrict__ src, half_t* __restrict__ dst, int n)
{
    int i = (blockIdx.x * 256 + threadIdx.x) * 8;
    if (i + 8 <= n) {
        v4f a = *(const v4f*)(src + i);
        v4f b = *(const v4f*)(src + i + 4);
        v8h h;
        #pragma unroll
        for (int j = 0; j < 4; ++j) { h[j] = (_Float16)a[j]; h[4 + j] = (_Float16)b[j]; }
        *(v8h*)(dst + i) = h;
    }
}

// ======================================================================
// Kernel 1: QKV projection, all-f16, async double-buffered LDS.
// Block 256 thr (8 waves), 128x64 tile; wave = 16 rows x 64 cols.
// ======================================================================
__global__ __launch_bounds__(256) void qkv_proj_kernel(
    const half_t* __restrict__ xh,
    const half_t* __restrict__ wqh,
    const half_t* __restrict__ wkh,
    const half_t* __restrict__ wvh,
    half_t* __restrict__ qkv)
{
    __shared__ _Float16 sA[2][BM * LDA_STR];
    __shared__ _Float16 sB[2][BN * LDB_STR];

    const int t    = threadIdx.x;
    const int lane = t & 31;
    const int wave = t >> 5;
    const int bn   = blockIdx.x * BN;
    const int bm   = blockIdx.y * BM;
    const int sel  = blockIdx.z;
    const half_t* W = (sel == 0) ? wqh : (sel == 1) ? wkh : wvh;
    half_t* out     = qkv + (size_t)sel * TOT;

    const int m16   = lane & 15;
    const int hsel  = lane >> 4;
    const int koffA = hsel << 4;
    const int kbB   = hsel << 4;

    const int ar0 = (t * 8) >> 5,         ac0 = (t * 8) & 31;
    const int ar1 = ((256 + t) * 8) >> 5, ac1 = ((256 + t) * 8) & 31;
    const int br  = t >> 2,               bc  = (t & 3) * 8;

    v8f acc[4] = {};

    auto issue = [&](int buf, int k0) {
        async_cp16(&sA[buf][ar0 * LDA_STR + ac0], xh + (size_t)(bm + ar0) * Cc + k0 + ac0);
        async_cp16(&sA[buf][ar1 * LDA_STR + ac1], xh + (size_t)(bm + ar1) * Cc + k0 + ac1);
        async_cp16(&sB[buf][br  * LDB_STR + bc ], W  + (size_t)(bn + br ) * Cc + k0 + bc );
    };

    issue(0, 0);
    for (int kt = 0; kt < NKT; ++kt) {
        const int cur = kt & 1;
        if (kt + 1 < NKT) { issue(1 - cur, (kt + 1) * BK); wait_async<3>(); }
        else              { wait_async<0>(); }
        __syncthreads();

        Frag a;
        {
            const _Float16* p = &sA[cur][(wave * 16 + m16) * LDA_STR + koffA];
            #pragma unroll
            for (int i = 0; i < 8; ++i) { a.e[i] = p[i]; a.e[8 + i] = p[8 + i]; }
        }
        #pragma unroll
        for (int j = 0; j < 4; ++j) {
            Frag bf;
            const _Float16* p = &sB[cur][(j * 16 + m16) * LDB_STR + kbB];
            #pragma unroll
            for (int i = 0; i < 16; ++i) bf.e[i] = p[i];
            acc[j] = wmma_f16(a.h, bf.h, acc[j]);
        }
        __syncthreads();
    }

    #pragma unroll
    for (int j = 0; j < 4; ++j) {
        int col = bn + j * 16 + m16;
        int hh = col >> 6, d = col & 63;
        #pragma unroll
        for (int r = 0; r < 8; ++r) {
            int m = bm + wave * 16 + r + hsel * 8;
            int bbx = m >> 11, nseq = m & (Nn - 1);
            out[(((size_t)bbx * Hh + hh) * Nn + nseq) * HD + d] = (_Float16)acc[j][r];
        }
    }
}

// ======================================================================
// Kernel 2: flash attention, async double-buffered K/V tiles.
// Block 128 thr (4 waves) = 64 queries for one (b, h); key tiles of 32.
// ======================================================================
#define KT 32
#define QT 64
#define KV_STR 72   // halves; 144 B = 9*16 -> b128-aligned rows
#define P_STR  40

__global__ __launch_bounds__(128) void attn_kernel(
    const half_t* __restrict__ qkv,
    half_t* __restrict__ o)
{
    __shared__ _Float16 sK[2][KT * KV_STR];
    __shared__ _Float16 sV[2][KT * KV_STR];
    __shared__ _Float16 sP[4][16 * P_STR];

    const int t    = threadIdx.x;
    const int lane = t & 31;
    const int wave = t >> 5;
    const int qt   = blockIdx.x * QT;
    const int hh   = blockIdx.y;
    const int bb   = blockIdx.z;

    const size_t bh = ((size_t)bb * Hh + hh) * Nn * HD;
    const half_t* Qp = qkv + bh;
    const half_t* Kp = qkv + (size_t)TOT + bh;
    const half_t* Vp = qkv + (size_t)2 * TOT + bh;

    const int m16  = lane & 15;
    const int hsel = lane >> 4;
    const int koff = hsel << 4;
    const int kb   = hsel << 4;

    const int cr0 = (t * 8) >> 6,         cc0 = (t * 8) & 63;
    const int cr1 = ((128 + t) * 8) >> 6, cc1 = ((128 + t) * 8) & 63;

    auto issue = [&](int buf, int k0) {
        async_cp16(&sK[buf][cr0 * KV_STR + cc0], Kp + (size_t)(k0 + cr0) * HD + cc0);
        async_cp16(&sK[buf][cr1 * KV_STR + cc1], Kp + (size_t)(k0 + cr1) * HD + cc1);
        async_cp16(&sV[buf][cr0 * KV_STR + cc0], Vp + (size_t)(k0 + cr0) * HD + cc0);
        async_cp16(&sV[buf][cr1 * KV_STR + cc1], Vp + (size_t)(k0 + cr1) * HD + cc1);
    };

    // Q fragments (2 A-frags cover HD = 64), loaded once
    Frag qa[2];
    {
        const half_t* qrow = Qp + (size_t)(qt + wave * 16 + m16) * HD;
        #pragma unroll
        for (int s = 0; s < 2; ++s)
            #pragma unroll
            for (int i = 0; i < 8; ++i) {
                qa[s].e[i]     = qrow[s * 32 + koff + i];
                qa[s].e[8 + i] = qrow[s * 32 + koff + 8 + i];
            }
    }

    float mrow[8], lrow[8];
    v8f oacc[4] = {};
    #pragma unroll
    for (int r = 0; r < 8; ++r) { mrow[r] = -1e30f; lrow[r] = 0.0f; }
    const float LOG2E = 1.44269504088896f;

    issue(0, 0);
    for (int kt = 0; kt < Nn / KT; ++kt) {
        const int cur = kt & 1;
        if (kt + 1 < Nn / KT) { issue(1 - cur, (kt + 1) * KT); wait_async<4>(); }
        else                  { wait_async<0>(); }
        __syncthreads();

        // S = Q.K^T : 16 queries x 32 keys (inner dim 64 -> 2 WMMA steps)
        v8f c0 = {}, c1 = {};
        #pragma unroll
        for (int s = 0; s < 2; ++s) {
            #pragma unroll
            for (int kk = 0; kk < 2; ++kk) {
                Frag bf;   // B[k=dim][n=key] = Ktile[key][dim]
                const _Float16* p = &sK[cur][(s * 16 + m16) * KV_STR + kk * 32 + kb];
                #pragma unroll
                for (int i = 0; i < 16; ++i) bf.e[i] = p[i];
                if (s == 0) c0 = wmma_f16(qa[kk].h, bf.h, c0);
                else        c1 = wmma_f16(qa[kk].h, bf.h, c1);
            }
        }
        #pragma unroll
        for (int r = 0; r < 8; ++r) { c0[r] *= 0.125f; c1[r] *= 0.125f; }

        // row-wise online softmax (rows live within 16-lane halves)
        float tmax[8], tsum[8];
        #pragma unroll
        for (int r = 0; r < 8; ++r) tmax[r] = fmaxf(c0[r], c1[r]);
        #pragma unroll
        for (int msk = 1; msk < 16; msk <<= 1)
            #pragma unroll
            for (int r = 0; r < 8; ++r)
                tmax[r] = fmaxf(tmax[r], __shfl_xor(tmax[r], msk, 32));

        #pragma unroll
        for (int r = 0; r < 8; ++r) {
            float mnew  = fmaxf(mrow[r], tmax[r]);
            float alpha = exp2f((mrow[r] - mnew) * LOG2E);
            mrow[r] = mnew;
            c0[r] = exp2f((c0[r] - mnew) * LOG2E);
            c1[r] = exp2f((c1[r] - mnew) * LOG2E);
            tsum[r] = c0[r] + c1[r];
            lrow[r] *= alpha;
            #pragma unroll
            for (int j = 0; j < 4; ++j) oacc[j][r] *= alpha;
        }
        #pragma unroll
        for (int msk = 1; msk < 16; msk <<= 1)
            #pragma unroll
            for (int r = 0; r < 8; ++r) tsum[r] += __shfl_xor(tsum[r], msk, 32);
        #pragma unroll
        for (int r = 0; r < 8; ++r) lrow[r] += tsum[r];

        // P: C-layout -> A-layout via wave-private LDS bounce
        _Float16* Pbuf = sP[wave];
        #pragma unroll
        for (int r = 0; r < 8; ++r) {
            int row = r + hsel * 8;
            Pbuf[row * P_STR + m16]      = (_Float16)c0[r];
            Pbuf[row * P_STR + 16 + m16] = (_Float16)c1[r];
        }
        Frag pa;
        {
            const _Float16* p = &Pbuf[m16 * P_STR + koff];
            #pragma unroll
            for (int i = 0; i < 8; ++i) { pa.e[i] = p[i]; pa.e[8 + i] = p[8 + i]; }
        }

        // O += P (16x32) x V (32x64)
        #pragma unroll
        for (int j = 0; j < 4; ++j) {
            Frag bf;   // B[k=key][n=dim]
            #pragma unroll
            for (int i = 0; i < 16; ++i)
                bf.e[i] = sV[cur][(kb + i) * KV_STR + j * 16 + m16];
            oacc[j] = wmma_f16(pa.h, bf.h, oacc[j]);
        }
        __syncthreads();
    }

    #pragma unroll
    for (int r = 0; r < 8; ++r) {
        float inv = 1.0f / lrow[r];
        int q = qt + wave * 16 + r + hsel * 8;
        half_t* orow = o + ((size_t)bb * Nn + q) * Cc + hh * HD;
        #pragma unroll
        for (int j = 0; j < 4; ++j)
            orow[j * 16 + m16] = (_Float16)(oacc[j][r] * inv);
    }
}

// ======================================================================
// Kernel 3: output projection, f16 A/W, f32 out + bias.
// ======================================================================
__global__ __launch_bounds__(256) void oproj_kernel(
    const half_t* __restrict__ A,
    const half_t* __restrict__ Wh,
    const float* __restrict__ bo,
    float* __restrict__ y)
{
    __shared__ _Float16 sA[2][BM * LDA_STR];
    __shared__ _Float16 sB[2][BN * LDB_STR];

    const int t    = threadIdx.x;
    const int lane = t & 31;
    const int wave = t >> 5;
    const int bn   = blockIdx.x * BN;
    const int bm   = blockIdx.y * BM;

    const int m16   = lane & 15;
    const int hsel  = lane >> 4;
    const int koffA = hsel << 4;
    const int kbB   = hsel << 4;

    const int ar0 = (t * 8) >> 5,         ac0 = (t * 8) & 31;
    const int ar1 = ((256 + t) * 8) >> 5, ac1 = ((256 + t) * 8) & 31;
    const int br  = t >> 2,               bc  = (t & 3) * 8;

    v8f acc[4] = {};

    auto issue = [&](int buf, int k0) {
        async_cp16(&sA[buf][ar0 * LDA_STR + ac0], A  + (size_t)(bm + ar0) * Cc + k0 + ac0);
        async_cp16(&sA[buf][ar1 * LDA_STR + ac1], A  + (size_t)(bm + ar1) * Cc + k0 + ac1);
        async_cp16(&sB[buf][br  * LDB_STR + bc ], Wh + (size_t)(bn + br ) * Cc + k0 + bc );
    };

    issue(0, 0);
    for (int kt = 0; kt < NKT; ++kt) {
        const int cur = kt & 1;
        if (kt + 1 < NKT) { issue(1 - cur, (kt + 1) * BK); wait_async<3>(); }
        else              { wait_async<0>(); }
        __syncthreads();

        Frag a;
        {
            const _Float16* p = &sA[cur][(wave * 16 + m16) * LDA_STR + koffA];
            #pragma unroll
            for (int i = 0; i < 8; ++i) { a.e[i] = p[i]; a.e[8 + i] = p[8 + i]; }
        }
        #pragma unroll
        for (int j = 0; j < 4; ++j) {
            Frag bf;
            const _Float16* p = &sB[cur][(j * 16 + m16) * LDB_STR + kbB];
            #pragma unroll
            for (int i = 0; i < 16; ++i) bf.e[i] = p[i];
            acc[j] = wmma_f16(a.h, bf.h, acc[j]);
        }
        __syncthreads();
    }

    #pragma unroll
    for (int j = 0; j < 4; ++j) {
        int col = bn + j * 16 + m16;
        float bias = bo[col];
        #pragma unroll
        for (int r = 0; r < 8; ++r) {
            int m = bm + wave * 16 + r + hsel * 8;
            y[(size_t)m * Cc + col] = acc[j][r] + bias;
        }
    }
}

// ======================================================================
extern "C" void kernel_launch(void* const* d_in, const int* in_sizes, int n_in,
                              void* d_out, int out_size, void* d_ws, size_t ws_size,
                              hipStream_t stream) {
    (void)in_sizes; (void)n_in; (void)out_size; (void)ws_size;

    const float* x  = (const float*)d_in[0];
    const float* Wq = (const float*)d_in[1];
    const float* Wk = (const float*)d_in[2];
    const float* Wv = (const float*)d_in[3];
    const float* Wo = (const float*)d_in[4];
    const float* bo = (const float*)d_in[5];
    float* y        = (float*)d_out;

    // ws layout (halves): xh | Wq | Wk | Wv | Wo | Q | K | V | O  (~36 MB)
    half_t* xh  = (half_t*)d_ws;
    half_t* wqh = xh  + (size_t)Mm * Cc;
    half_t* wkh = wqh + WSZ;
    half_t* wvh = wkh + WSZ;
    half_t* woh = wvh + WSZ;
    half_t* qkv = woh + WSZ;
    half_t* obuf = qkv + (size_t)3 * TOT;

    cvt_kernel<<<(Mm * Cc) / 2048, 256, 0, stream>>>(x,  xh,  Mm * Cc);
    cvt_kernel<<<WSZ / 2048, 256, 0, stream>>>(Wq, wqh, WSZ);
    cvt_kernel<<<WSZ / 2048, 256, 0, stream>>>(Wk, wkh, WSZ);
    cvt_kernel<<<WSZ / 2048, 256, 0, stream>>>(Wv, wvh, WSZ);
    cvt_kernel<<<WSZ / 2048, 256, 0, stream>>>(Wo, woh, WSZ);

    dim3 g1(Cc / BN, Mm / BM, 3);
    qkv_proj_kernel<<<g1, 256, 0, stream>>>(xh, wqh, wkh, wvh, qkv);

    dim3 g2(Nn / QT, Hh, Bb);
    attn_kernel<<<g2, 128, 0, stream>>>(qkv, obuf);

    dim3 g3(Cc / BN, Mm / BM);
    oproj_kernel<<<g3, 256, 0, stream>>>(obuf, woh, bo, y);
}
